// DirectedWeightedGraphEmbedding_77489799954660
// MI455X (gfx1250) — compile-verified, matrix operand order checked by reading, not run on hardware
//
#include <hip/hip_runtime.h>
#include <hip/hip_bf16.h>

typedef __attribute__((ext_vector_type(16))) __bf16 v16bf;
typedef __attribute__((ext_vector_type(8)))  float  v8f;

#define HIDF 128
#define ENCF 64

// ---------------- degree / graph kernels ----------------

__global__ void k_deg_init(float* deg, int n) {
  int i = blockIdx.x * blockDim.x + threadIdx.x;
  if (i < n) deg[i] = 1.0f;                       // self-loop weight
}

__global__ void k_deg_accum(const int* __restrict__ dst, const float* __restrict__ w,
                            float* deg, int e) {
  int i = blockIdx.x * blockDim.x + threadIdx.x;
  if (i < e) atomicAdd(&deg[__builtin_nontemporal_load(dst + i)],
                       __builtin_nontemporal_load(w + i));
}

__global__ void k_rsqrt(float* deg, int n) {
  int i = blockIdx.x * blockDim.x + threadIdx.x;
  if (i < n) deg[i] = rsqrtf(deg[i]);             // deg >= 1 always (self loop)
}

// one wave per edge: lane l handles features 4l..4l+3 (float4 gather, 4 atomics).
// Edge-list reads are non-temporal (streamed once) so h/out stay L2-resident.
__global__ void k_edge_scatter(const int* __restrict__ src, const int* __restrict__ dst,
                               const float* __restrict__ w, const float* __restrict__ dinv,
                               const float* __restrict__ h, float* __restrict__ out, int e) {
  int wid  = (int)((blockIdx.x * (size_t)blockDim.x + threadIdx.x) >> 5);
  int lane = threadIdx.x & 31;
  if (wid >= e) return;
  int s = __builtin_nontemporal_load(src + wid);
  int d = __builtin_nontemporal_load(dst + wid);
  float we = __builtin_nontemporal_load(w + wid);
  float norm = dinv[s] * we * dinv[d];
  const float4* hs = (const float4*)(h + (size_t)s * HIDF);
  float4 v = hs[lane];
  float* od = out + (size_t)d * HIDF + lane * 4;
  atomicAdd(od + 0, v.x * norm);
  atomicAdd(od + 1, v.y * norm);
  atomicAdd(od + 2, v.z * norm);
  atomicAdd(od + 3, v.w * norm);
}

// ---------------- fused WMMA bf16 GEMM ----------------
//  A-path (optional): x <- relu(x + in_bias[k])       (fuses previous layer's bias+ReLU)
//  Y[n,OUT] = A @ W[OUT,K]^T (+ out_bias)
//  ACC path (optional): ACC[n,OUT] = Y * dinv[n]^2    (fuses self-loop accumulator init)
// One block = 16 rows, OUT/16 waves; wave w owns the 16x16 tile at col 16w.
template <int K, int OUT, bool IN_ACT, bool WRITE_ACC, bool OUT_BIAS>
__global__ __launch_bounds__(32 * (OUT / 16))
void k_gemm(const float* X, const float* __restrict__ W,
            const float* __restrict__ in_bias,   // [K]   if IN_ACT
            const float* __restrict__ out_bias,  // [OUT] if OUT_BIAS
            const float* __restrict__ dinv,      // [n]   if WRITE_ACC
            float* __restrict__ Y,               // [n,OUT]
            float* ACC,                          // [n,OUT] (may alias X; per-block row-safe)
            int n_rows) {
  __shared__ __bf16 Wl[OUT * K];                 // bf16 weights in LDS (<=32KB)
  __shared__ float  Bl[K];
  const int nthreads = 32 * (OUT / 16);
  for (int i = threadIdx.x; i < OUT * K; i += nthreads) Wl[i] = (__bf16)W[i];
  if (IN_ACT)
    for (int i = threadIdx.x; i < K; i += nthreads) Bl[i] = in_bias[i];
  __syncthreads();

  const int wave = threadIdx.x >> 5;
  const int lane = threadIdx.x & 31;
  const int half = lane >> 4;                    // 0: lanes 0-15, 1: lanes 16-31
  const int r    = lane & 15;
  const int m0   = blockIdx.x * 16;
  const int o0   = wave * 16;

  int rowA = m0 + r; if (rowA >= n_rows) rowA = n_rows - 1;   // clamp (loads only)
  const float*  xrow = X  + (size_t)rowA * K;
  const __bf16* wrow = Wl + (size_t)(o0 + r) * K;             // B col n -> W row o0+n

  v8f acc = {};
  for (int k0 = 0; k0 < K; k0 += 32) {
    v16bf a, b;
#pragma unroll
    for (int j = 0; j < 8; ++j) {
      // 16-bit A layout: lanes 0-15 hold K {0..7,16..23}; lanes 16-31 hold K {8..15,24..31}
      int kA = k0 + 8 * half + j;
      int kB = k0 + 16 + 8 * half + j;
      float x0 = xrow[kA], x1 = xrow[kB];
      if (IN_ACT) {
        x0 += Bl[kA]; x0 = x0 > 0.0f ? x0 : 0.0f;
        x1 += Bl[kB]; x1 = x1 > 0.0f ? x1 : 0.0f;
      }
      a[j]     = (__bf16)x0;
      a[j + 8] = (__bf16)x1;
      b[j]     = wrow[kA - k0 + k0];             // == wrow[kA]
      b[j + 8] = wrow[kB];
    }
    acc = __builtin_amdgcn_wmma_f32_16x16x32_bf16(
        /*neg_a=*/false, a, /*neg_b=*/false, b,
        /*c_mod=*/(short)0, acc, /*reuse_a=*/false, /*reuse_b=*/false);
  }

  // C/D layout: VGPR j -> row m0 + j + 8*half, col o0 + (lane&15)
  const int col = o0 + r;
#pragma unroll
  for (int j = 0; j < 8; ++j) {
    int row = m0 + j + 8 * half;
    if (row < n_rows) {
      float v = acc[j];
      if (OUT_BIAS) v += out_bias[col];
      Y[(size_t)row * OUT + col] = v;
      if (WRITE_ACC) {
        float di = dinv[row];
        ACC[(size_t)row * OUT + col] = v * di * di;
      }
    }
  }
}

// ---------------- driver ----------------

extern "C" void kernel_launch(void* const* d_in, const int* in_sizes, int n_in,
                              void* d_out, int out_size, void* d_ws, size_t ws_size,
                              hipStream_t stream) {
  const float* x  = (const float*)d_in[0];
  const int*   ei = (const int*)d_in[1];
  const float* ew = (const float*)d_in[2];
  const float* W1 = (const float*)d_in[3];
  const float* b1 = (const float*)d_in[4];
  const float* W2 = (const float*)d_in[5];
  const float* b2 = (const float*)d_in[6];
  const float* Wf = (const float*)d_in[7];
  const float* bf = (const float*)d_in[8];
  float* out = (float*)d_out;

  const int N = in_sizes[0] / HIDF;
  const int E = in_sizes[1] / 2;
  const int* src = ei;
  const int* dst = ei + E;

  // workspace layout: dinv[N] | bufA[N*128] (T) | bufB[N*128] (ACC / next-layer input)
  float* dinv = (float*)d_ws;
  size_t off  = ((size_t)N + 255) & ~(size_t)255;
  float* bufA = dinv + off;
  float* bufB = bufA + (size_t)N * HIDF;

  const int nodeTiles = (N + 15) / 16;

  // weighted in-degree (incl. self loop) -> dinv
  k_deg_init <<<(N + 255) / 256, 256, 0, stream>>>(dinv, N);
  k_deg_accum<<<(E + 255) / 256, 256, 0, stream>>>(dst, ew, dinv, E);
  k_rsqrt    <<<(N + 255) / 256, 256, 0, stream>>>(dinv, N);

  // ----- layer 1: T1 = x@W1^T -> bufA;  bufB = T1*dinv^2;  scatter edges into bufB -----
  k_gemm<HIDF, HIDF, false, true, false><<<nodeTiles, 256, 0, stream>>>(
      x, W1, nullptr, nullptr, dinv, bufA, bufB, N);
  k_edge_scatter<<<(E + 7) / 8, 256, 0, stream>>>(src, dst, ew, dinv, bufA, bufB, E);

  // ----- layer 2: A = relu(bufB + b1); T2 -> bufA; bufB = T2*dinv^2 (in place); scatter -----
  k_gemm<HIDF, HIDF, true, true, false><<<nodeTiles, 256, 0, stream>>>(
      bufB, W2, b1, nullptr, dinv, bufA, bufB, N);
  k_edge_scatter<<<(E + 7) / 8, 256, 0, stream>>>(src, dst, ew, dinv, bufA, bufB, E);

  // ----- final: out = relu(bufB + b2) @ Wf^T + bf -----
  k_gemm<HIDF, ENCF, true, false, true><<<nodeTiles, 128, 0, stream>>>(
      bufB, Wf, b2, bf, nullptr, out, nullptr, N);
}
